// Attention_43301860278716
// MI455X (gfx1250) — matrix operand order repaired and checked by reading, not compiled, not run on hardware
//
#include <hip/hip_runtime.h>
#include <hip/hip_bf16.h>

// ---------------------------------------------------------------------------
// Problem constants (from reference)
// ---------------------------------------------------------------------------
#define BB    2
#define SS    1024
#define HID   4096
#define NH    32
#define NKV   8
#define DD    128
#define MTOK  (BB * SS)          // 2048 tokens
#define EPSF  1e-6f

typedef __attribute__((ext_vector_type(16))) __bf16 v16bf;
typedef __attribute__((ext_vector_type(8)))  __bf16 v8bf;
typedef __attribute__((ext_vector_type(8)))  float  v8f;

#define WMMA_BF16(A, B, C)                                                    \
  __builtin_amdgcn_wmma_f32_16x16x32_bf16(false, (A), false, (B), (short)0,   \
                                          (C), false, false)

// float -> bf16 (round-to-nearest-even), bit level
__device__ __forceinline__ __bf16 f2bf(float f) {
  unsigned int u = __builtin_bit_cast(unsigned int, f);
  u += 0x7FFFu + ((u >> 16) & 1u);
  unsigned short s = (unsigned short)(u >> 16);
  return __builtin_bit_cast(__bf16, s);
}

// ---------------------------------------------------------------------------
// Async global -> LDS copy (16 bytes per lane), CDNA5 ASYNCcnt path.
// Builtin signature (from compiler diagnostic): param0 = AS(1) int4*,
// param1 = AS(3) int4*, then imm offset + imm cpol.
// ---------------------------------------------------------------------------
typedef int i32x4 __attribute__((vector_size(16)));
typedef __attribute__((address_space(1))) i32x4 gi32x4;
typedef __attribute__((address_space(3))) i32x4 li32x4;

__device__ __forceinline__ void cp_async16(const __bf16* g, __bf16* l) {
#if __has_builtin(__builtin_amdgcn_global_load_async_to_lds_b128)
  __builtin_amdgcn_global_load_async_to_lds_b128((gi32x4*)g, (li32x4*)l, 0, 0);
#else
  *(v8bf*)l = *(const v8bf*)g;
#endif
}

__device__ __forceinline__ void wait_async0() {
#if __has_builtin(__builtin_amdgcn_s_wait_asynccnt)
  __builtin_amdgcn_s_wait_asynccnt(0);
#else
  asm volatile("s_wait_asynccnt 0" ::: "memory");
#endif
}

// ---------------------------------------------------------------------------
// Kernel 0: f32 -> bf16 convert (grid-stride)
// ---------------------------------------------------------------------------
__global__ void cvt_f32_bf16(const float* __restrict__ in,
                             __bf16* __restrict__ out, int n) {
  int i = blockIdx.x * blockDim.x + threadIdx.x;
  int stride = gridDim.x * blockDim.x;
  for (; i < n; i += stride) out[i] = f2bf(in[i]);
}

// ---------------------------------------------------------------------------
// Kernel 1: C[M,N] = A[M,K] @ W[N,K]^T  (bf16 in, f32 out)
// 256 threads = 8 waves arranged 2(M) x 4(N); each wave owns a 32x64 tile
// (8 accumulators). K-loop is software-pipelined with explicit register
// double buffering so WMMAs on tile k overlap loads of tile k+32.
//
// Fragment layouts (A and B share a contiguous k order, not the ISA table order):
//  A (16x32 bf16): lane = ln + 16*hi, row M = ln,
//    elems 0..15 = K[16*hi..+15]
//  B (32x16 bf16): column N = ln, elems 0..15 = K[16*hi..+15]
//  C (16x16 f32): vgpr r, lanes0-15 -> (M=r,N=ln); lanes16-31 -> (M=8+r,N=ln)
// ---------------------------------------------------------------------------
__launch_bounds__(256)
__global__ void gemm_bf16_wmma(const __bf16* __restrict__ A,
                               const __bf16* __restrict__ W,
                               float* __restrict__ C, int M, int N, int K) {
  const int lane = threadIdx.x & 31;
  const int wave = threadIdx.x >> 5;
  const int ln = lane & 15;
  const int hi = lane >> 4;
  const int wm = wave >> 2;  // 0..1
  const int wn = wave & 3;   // 0..3
  const int m0 = blockIdx.y * 64 + wm * 32;
  const int n0 = blockIdx.x * 256 + wn * 64;

  union AF { v16bf v; v8bf h2[2]; };
  struct Frags {
    AF a0, a1;
    v16bf b[4];
  };

  v8f acc[2][4];
#pragma unroll
  for (int mi = 0; mi < 2; ++mi)
#pragma unroll
    for (int j = 0; j < 4; ++j)
#pragma unroll
      for (int r = 0; r < 8; ++r) acc[mi][j][r] = 0.0f;

  const __bf16* ar0 = A + (size_t)(m0 + ln) * K;
  const __bf16* ar1 = A + (size_t)(m0 + 16 + ln) * K;

  Frags f0, f1;
  auto loadFrags = [&](Frags& f, int k0) {
    f.a0.h2[0] = *(const v8bf*)(ar0 + k0 + 16 * hi);
    f.a0.h2[1] = *(const v8bf*)(ar0 + k0 + 16 * hi + 8);
    f.a1.h2[0] = *(const v8bf*)(ar1 + k0 + 16 * hi);
    f.a1.h2[1] = *(const v8bf*)(ar1 + k0 + 16 * hi + 8);
#pragma unroll
    for (int j = 0; j < 4; ++j)
      f.b[j] =
          *(const v16bf*)(W + (size_t)(n0 + j * 16 + ln) * K + k0 + 16 * hi);
    if (k0 + 256 < K) {
      __builtin_prefetch(ar0 + k0 + 256, 0, 3);
      __builtin_prefetch(ar1 + k0 + 256, 0, 3);
    }
  };
  auto mmaFrags = [&](Frags& f) {
#pragma unroll
    for (int j = 0; j < 4; ++j) {
      acc[0][j] = WMMA_BF16(f.a0.v, f.b[j], acc[0][j]);
      acc[1][j] = WMMA_BF16(f.a1.v, f.b[j], acc[1][j]);
    }
  };

  loadFrags(f0, 0);
  for (int k0 = 0; k0 < K; k0 += 64) {   // K is a multiple of 64
    if (k0 + 32 < K) loadFrags(f1, k0 + 32);
    mmaFrags(f0);
    if (k0 + 64 < K) loadFrags(f0, k0 + 64);
    mmaFrags(f1);
  }

#pragma unroll
  for (int mi = 0; mi < 2; ++mi)
#pragma unroll
    for (int j = 0; j < 4; ++j)
#pragma unroll
      for (int r = 0; r < 8; ++r) {
        int row = m0 + mi * 16 + hi * 8 + r;
        int col = n0 + j * 16 + ln;
        C[(size_t)row * N + col] = acc[mi][j][r];
      }
}

// ---------------------------------------------------------------------------
// Kernel 2: fused per-head RMSNorm + RoPE, f32 [tok, nh*D] -> bf16 [b,h,S,D]
// ---------------------------------------------------------------------------
__launch_bounds__(128)
__global__ void norm_rope_kernel(const float* __restrict__ X,
                                 const float* __restrict__ w,
                                 const float* __restrict__ freqs,
                                 __bf16* __restrict__ out, int nh) {
  const int bid = blockIdx.x;
  const int h = bid % nh;
  const int t = bid / nh;  // token = b*S + s
  const int s = t % SS;
  const int b = t / SS;
  const int d = threadIdx.x;

  float x = X[(size_t)t * nh * DD + h * DD + d];

  __shared__ float red[128];
  __shared__ float xn_s[128];
  red[d] = x * x;
  __syncthreads();
#pragma unroll
  for (int off = 64; off > 0; off >>= 1) {
    if (d < off) red[d] += red[d + off];
    __syncthreads();
  }
  float rms = rsqrtf(red[0] / (float)DD + EPSF);
  float xn = x * rms * w[d];
  xn_s[d] = xn;
  __syncthreads();

  const float* cosp = freqs + (size_t)s * DD;
  const float* sinp = freqs + (size_t)SS * DD + (size_t)s * DD;
  const float* nsinp = freqs + 2 * (size_t)SS * DD + (size_t)s * DD;

  float e = xn * cosp[d];
  float r;
  if (d < 64)
    r = e + xn_s[d + 64] * nsinp[d];
  else
    r = e + xn_s[d - 64] * sinp[d];

  out[(((size_t)b * nh + h) * SS + s) * DD + d] = f2bf(r);
}

// ---------------------------------------------------------------------------
// Kernel 3: V f32 [tok, KV*D] -> bf16 transposed [b, kv, D, S]
// ---------------------------------------------------------------------------
__global__ void v_transpose_kernel(const float* __restrict__ V,
                                   __bf16* __restrict__ Vt) {
  size_t i = (size_t)blockIdx.x * blockDim.x + threadIdx.x;
  size_t total = (size_t)MTOK * NKV * DD;
  size_t stride = (size_t)gridDim.x * blockDim.x;
  for (; i < total; i += stride) {
    int d = (int)(i % DD);
    size_t r = i / DD;
    int h = (int)(r % NKV);
    size_t t = r / NKV;
    int s = (int)(t % SS);
    int b = (int)(t / SS);
    Vt[(((size_t)b * NKV + h) * DD + d) * SS + s] = f2bf(V[i]);
  }
}

// ---------------------------------------------------------------------------
// Kernel 4: GQA flash attention.
// Block = 4 waves, all on the SAME (b, head): 64 consecutive q rows.
// K/V 32-key tiles are cooperatively staged in LDS (shared by all 4 waves,
// 4x less global traffic), double-buffered via async global->LDS copies
// tracked with ASYNCcnt. QK^T and P@V via v_wmma_f32_16x16x32_bf16.
// ---------------------------------------------------------------------------
#define KPAD (DD + 8)   // 272B rows -> conflict-free across ln lanes
#define VPAD (32 + 8)   // 80B rows

__launch_bounds__(128)
__global__ void flash_attn_wmma(const __bf16* __restrict__ Q,
                                const __bf16* __restrict__ Km,
                                const __bf16* __restrict__ Vt,
                                __bf16* __restrict__ O) {
  const int tid = threadIdx.x;
  const int lane = tid & 31;
  const int wave = tid >> 5;
  const int ln = lane & 15;
  const int hi = lane >> 4;

  const int grpPerBH = SS / 64;  // 16
  const int qg = blockIdx.x % grpPerBH;
  const int bh = blockIdx.x / grpPerBH;
  const int h = bh % NH;
  const int b = bh / NH;
  const int kvh = h / (NH / NKV);
  const int q0 = qg * 64 + wave * 16;

  const __bf16* qbase = Q + (((size_t)b * NH + h) * SS + q0) * DD;
  const __bf16* kbase = Km + (((size_t)b * NKV + kvh) * SS) * DD;
  const __bf16* vbase = Vt + (((size_t)b * NKV + kvh) * DD) * SS;

  __shared__ __bf16 Kt[2][32][KPAD];   // [key][d]
  __shared__ __bf16 Vs[2][DD][VPAD];   // [d][key]
  __shared__ __bf16 Pb[4][16][32];
  __bf16(*P)[32] = Pb[wave];

  union AF { v16bf v; v8bf h2[2]; };

  // cooperative async staging of one 32-key K/V tile
  auto stageTile = [&](int buf, int kb) {
#pragma unroll
    for (int i = 0; i < 4; ++i) {               // K: 512 x 16B chunks
      int chunk = i * 128 + tid;
      int key = chunk >> 4, c = chunk & 15;
      cp_async16(kbase + (size_t)(kb + key) * DD + c * 8, &Kt[buf][key][c * 8]);
    }
#pragma unroll
    for (int i = 0; i < 4; ++i) {               // V: 512 x 16B chunks
      int chunk = i * 128 + tid;
      int dd = chunk >> 2, c = chunk & 3;
      cp_async16(vbase + (size_t)dd * SS + kb + c * 8, &Vs[buf][dd][c * 8]);
    }
  };

  // Q A-fragments for the 4 K-slices of D=128 (kept in registers)
  AF qf[4];
#pragma unroll
  for (int t = 0; t < 4; ++t) {
    const __bf16* qr = qbase + (size_t)ln * DD + t * 32;
    qf[t].h2[0] = *(const v8bf*)(qr + 16 * hi);
    qf[t].h2[1] = *(const v8bf*)(qr + 16 * hi + 8);
  }

  v8f accO[8];
#pragma unroll
  for (int j = 0; j < 8; ++j)
#pragma unroll
    for (int r = 0; r < 8; ++r) accO[j][r] = 0.0f;

  float mrow[8], lrow[8];
#pragma unroll
  for (int r = 0; r < 8; ++r) { mrow[r] = -1e30f; lrow[r] = 0.0f; }

  const float scale = 0.0883883476483184f;  // 1/sqrt(128)

  stageTile(0, 0);
  wait_async0();
  __syncthreads();

  int kb = 0;
  for (int it = 0; it < SS / 32; ++it, kb += 32) {
    const int buf = it & 1;
    if (kb + 32 < SS) stageTile(buf ^ 1, kb + 32);

    // ---- scores: two 16x16 tiles over keys kb..kb+31, K operand from LDS --
    v8f sA, sB;
#pragma unroll
    for (int r = 0; r < 8; ++r) { sA[r] = 0.0f; sB[r] = 0.0f; }
#pragma unroll
    for (int t = 0; t < 4; ++t) {
      AF kfA, kfB;
      const __bf16* ka = &Kt[buf][ln][t * 32 + 16 * hi];
      const __bf16* kc = &Kt[buf][16 + ln][t * 32 + 16 * hi];
      kfA.h2[0] = *(const v8bf*)ka;
      kfA.h2[1] = *(const v8bf*)(ka + 8);
      kfB.h2[0] = *(const v8bf*)kc;
      kfB.h2[1] = *(const v8bf*)(kc + 8);
      sA = WMMA_BF16(qf[t].v, kfA.v, sA);
      sB = WMMA_BF16(qf[t].v, kfB.v, sB);
    }

    // ---- online softmax (row R = 8*hi + r lives in this lane's half) ----
    float alpha[8];
#pragma unroll
    for (int r = 0; r < 8; ++r) {
      float a = sA[r] * scale;
      float c = sB[r] * scale;
      float mx = fmaxf(a, c);
      mx = fmaxf(mx, __shfl_xor(mx, 1));
      mx = fmaxf(mx, __shfl_xor(mx, 2));
      mx = fmaxf(mx, __shfl_xor(mx, 4));
      mx = fmaxf(mx, __shfl_xor(mx, 8));
      float mnew = fmaxf(mrow[r], mx);
      alpha[r] = __expf(mrow[r] - mnew);
      mrow[r] = mnew;
      float pa = __expf(a - mnew);
      float pc = __expf(c - mnew);
      float sum = pa + pc;
      sum += __shfl_xor(sum, 1);
      sum += __shfl_xor(sum, 2);
      sum += __shfl_xor(sum, 4);
      sum += __shfl_xor(sum, 8);
      lrow[r] = lrow[r] * alpha[r] + sum;
      P[hi * 8 + r][ln] = f2bf(pa);
      P[hi * 8 + r][16 + ln] = f2bf(pc);
    }
    asm volatile("s_wait_dscnt 0" ::: "memory");

    // ---- reload P as a 16x32 A-fragment ----
    AF pf;
    pf.h2[0] = *(const v8bf*)(&P[ln][16 * hi]);
    pf.h2[1] = *(const v8bf*)(&P[ln][16 * hi + 8]);

    // ---- rescale accumulator, then P @ V (V operand from LDS) ----
#pragma unroll
    for (int j = 0; j < 8; ++j) {
#pragma unroll
      for (int r = 0; r < 8; ++r) accO[j][r] *= alpha[r];
      AF vf;
      const __bf16* vp = &Vs[buf][j * 16 + ln][16 * hi];
      vf.h2[0] = *(const v8bf*)vp;
      vf.h2[1] = *(const v8bf*)(vp + 8);
      accO[j] = WMMA_BF16(pf.v, vf.v, accO[j]);
    }

    wait_async0();     // next tile fully in LDS (this wave's share)
    __syncthreads();   // everyone done computing on buf + staging buf^1
  }

  // ---- epilogue: normalize, store bf16 [tok, H*D] for the Wo GEMM ----
#pragma unroll
  for (int j = 0; j < 8; ++j)
#pragma unroll
    for (int r = 0; r < 8; ++r) {
      int srow = q0 + hi * 8 + r;
      size_t tok = (size_t)b * SS + srow;
      float inv = (lrow[r] > 0.0f) ? (1.0f / lrow[r]) : 0.0f;
      O[tok * (NH * DD) + h * DD + j * 16 + ln] = f2bf(accO[j][r] * inv);
    }
}

// ---------------------------------------------------------------------------
// Launch
// ---------------------------------------------------------------------------
extern "C" void kernel_launch(void* const* d_in, const int* in_sizes, int n_in,
                              void* d_out, int out_size, void* d_ws,
                              size_t ws_size, hipStream_t stream) {
  const float* hidden = (const float*)d_in[0];
  const float* freqs = (const float*)d_in[1];
  const float* Wq = (const float*)d_in[2];
  const float* Wk = (const float*)d_in[3];
  const float* Wv = (const float*)d_in[4];
  const float* Wo = (const float*)d_in[5];
  const float* qnw = (const float*)d_in[6];
  const float* knw = (const float*)d_in[7];
  float* out = (float*)d_out;

  char* ws = (char*)d_ws;
  size_t off = 0;
  auto carve = [&](size_t bytes) -> char* {
    char* p = ws + off;
    off = (off + bytes + 255) & ~(size_t)255;
    return p;
  };
  __bf16* hidb = (__bf16*)carve((size_t)MTOK * HID * 2);
  __bf16* wqb  = (__bf16*)carve((size_t)NH * DD * HID * 2);
  __bf16* wkb  = (__bf16*)carve((size_t)NKV * DD * HID * 2);
  __bf16* wvb  = (__bf16*)carve((size_t)NKV * DD * HID * 2);
  __bf16* wob  = (__bf16*)carve((size_t)HID * NH * DD * 2);
  float*  qf   = (float*)carve((size_t)MTOK * NH * DD * 4);
  float*  kf   = (float*)carve((size_t)MTOK * NKV * DD * 4);
  float*  vf   = (float*)carve((size_t)MTOK * NKV * DD * 4);
  __bf16* qb   = (__bf16*)carve((size_t)MTOK * NH * DD * 2);
  __bf16* kb   = (__bf16*)carve((size_t)MTOK * NKV * DD * 2);
  __bf16* vtb  = (__bf16*)carve((size_t)MTOK * NKV * DD * 2);
  __bf16* attb = (__bf16*)carve((size_t)MTOK * NH * DD * 2);

  // 0) convert activations + weights to bf16
  cvt_f32_bf16<<<4096, 256, 0, stream>>>(hidden, hidb, MTOK * HID);
  cvt_f32_bf16<<<4096, 256, 0, stream>>>(Wq, wqb, NH * DD * HID);
  cvt_f32_bf16<<<4096, 256, 0, stream>>>(Wk, wkb, NKV * DD * HID);
  cvt_f32_bf16<<<4096, 256, 0, stream>>>(Wv, wvb, NKV * DD * HID);
  cvt_f32_bf16<<<4096, 256, 0, stream>>>(Wo, wob, HID * NH * DD);

  // 1) projection GEMMs (block tile 64M x 256N)
  gemm_bf16_wmma<<<dim3((NH * DD) / 256, MTOK / 64), 256, 0, stream>>>(
      hidb, wqb, qf, MTOK, NH * DD, HID);
  gemm_bf16_wmma<<<dim3((NKV * DD) / 256, MTOK / 64), 256, 0, stream>>>(
      hidb, wkb, kf, MTOK, NKV * DD, HID);
  gemm_bf16_wmma<<<dim3((NKV * DD) / 256, MTOK / 64), 256, 0, stream>>>(
      hidb, wvb, vf, MTOK, NKV * DD, HID);

  // 2) per-head RMSNorm + RoPE (Q, K); V convert + transpose
  norm_rope_kernel<<<MTOK * NH, 128, 0, stream>>>(qf, qnw, freqs, qb, NH);
  norm_rope_kernel<<<MTOK * NKV, 128, 0, stream>>>(kf, knw, freqs, kb, NKV);
  v_transpose_kernel<<<4096, 256, 0, stream>>>(vf, vtb);

  // 3) flash attention: one block per (b, h, 64 q rows)
  flash_attn_wmma<<<BB * NH * (SS / 64), 128, 0, stream>>>(qb, kb, vtb, attb);

  // 4) output projection -> d_out (f32)
  gemm_bf16_wmma<<<dim3(HID / 256, MTOK / 64), 256, 0, stream>>>(
      attb, wob, out, MTOK, HID, NH * DD);
}
